// GPNNCell_20641612824993
// MI455X (gfx1250) — matrix operand order repaired and checked by reading, not compiled, not run on hardware
//
#include <hip/hip_runtime.h>
#include <math.h>

// ---- types for WMMA ----
typedef __attribute__((ext_vector_type(16))) _Float16 v16h;
typedef __attribute__((ext_vector_type(8)))  _Float16 v8h;
typedef __attribute__((ext_vector_type(4)))  _Float16 v4h;
typedef __attribute__((ext_vector_type(8)))  float    v8f;

#define DD 768
#define HH 256

// LDS layout (bytes)
#define OFF_E    0         // f16 [64][768]  = 98304
#define OFF_H    98304     // f16 [64][256]  = 32768
#define OFF_MV   196608    // f16 [64][768]  = 98304  (ends 294912)
#define OFF_M    0         // f32 [64][768]  = 196608 (phase3+, overlaps E/H)
#define OFF_WT   294912    // f32 [64]
#define OFF_STAT 295168    // f32 [64][4][2]
#define OFF_MUR  297216    // f32 [64][2]
#define SMEM_SZ  297728

// ws layout (bytes)
#define WS_WGT   0          // f16 [1024][768]  W_gates^T
#define WS_WMB   1572864    // f16 [768][768]   W_msg(bottom)^T
#define WS_WMG   2752512    // f16 [768][768]   W_mrg^T
#define WS_P     3932160    // f32 [512][768]   node@W_msg_top + b_msg
// total 5505024 bytes of workspace

// ---- branchless fast transcendentals (hardware v_exp_f32 path, no libm slow paths) ----
__device__ __forceinline__ float sigmoidf_(float x) {
    return __fdividef(1.0f, 1.0f + __expf(-x));
}
__device__ __forceinline__ float tanhf_(float x) {
    // 1 - 2/(e^{2x}+1): saturates to +/-1 at the extremes without branches
    return 1.0f - __fdividef(2.0f, __expf(2.0f * x) + 1.0f);
}
__device__ __forceinline__ float erff_(float x) {
    // Abramowitz-Stegun 7.1.26, max abs error ~1.5e-7
    float ax = fabsf(x);
    float t  = __fdividef(1.0f, fmaf(0.3275911f, ax, 1.0f));
    float p  = t * fmaf(t, fmaf(t, fmaf(t, fmaf(t, 1.061405429f, -1.453152027f),
                                        1.421413741f), -0.284496736f), 0.254829592f);
    float y  = 1.0f - p * __expf(-ax * ax);
    return copysignf(y, x);
}
__device__ __forceinline__ float gelu_erf_(float x) {
    return 0.5f * x * (1.0f + erff_(x * 0.70710678118654752f));
}

// Build a 16x32 f16 A-fragment for lane `lane` from row-major LDS matrix (stride 768 halves).
// A layout (contiguous k, matching B): lanes 0-15 hold K=[kb..kb+15]; lanes 16-31 hold K=[kb+16..kb+31].
__device__ __forceinline__ v16h load_afrag(const _Float16* base, int row, int kb, int halfsel) {
    const _Float16* p = base + row * DD + kb + 16 * halfsel;
    v8h a0 = *(const v8h*)(p);
    v8h a1 = *(const v8h*)(p + 8);
    return __builtin_shufflevector(a0, a1, 0, 1, 2, 3, 4, 5, 6, 7, 8, 9, 10, 11, 12, 13, 14, 15);
}

// ---------------- prep kernels ----------------

__global__ void prep_weights(const float* __restrict__ Wg, const float* __restrict__ Wmsg,
                             const float* __restrict__ Wmrg, _Float16* __restrict__ WgT,
                             _Float16* __restrict__ WmbT, _Float16* __restrict__ WmgT) {
    int idx = blockIdx.x * 256 + threadIdx.x;
    const int n1 = 1024 * DD, n2 = DD * DD;
    if (idx < n1) {
        int n = idx / DD, k = idx % DD;
        WgT[idx] = (_Float16)Wg[k * 1024 + n];
    } else if (idx < n1 + n2) {
        int j = idx - n1; int n = j / DD, k = j % DD;
        WmbT[j] = (_Float16)Wmsg[(DD + k) * DD + n];          // bottom half of W_msg
    } else {
        int j = idx - n1 - n2; int n = j / DD, k = j % DD;
        WmgT[j] = (_Float16)Wmrg[k * DD + n];
    }
}

// P[row=(b*64+w)][d] = node[row]@W_msg_top[:,d] + b_msg[d]   (tiny: 0.6 GFLOP)
__global__ void prep_P(const float* __restrict__ node, const float* __restrict__ Wmsg,
                       const float* __restrict__ b_msg, float* __restrict__ P) {
    __shared__ float nrow[DD];
    int row = blockIdx.x, tid = threadIdx.x;
    for (int i = tid; i < DD; i += 256) nrow[i] = node[(size_t)row * DD + i];
    __syncthreads();
    for (int j = 0; j < 3; ++j) {
        int d = tid + j * 256;
        float acc = b_msg[d];
        for (int k = 0; k < DD; ++k) acc = fmaf(nrow[k], Wmsg[(size_t)k * DD + d], acc);
        P[(size_t)row * DD + d] = acc;
    }
}

// ---------------- main fused kernel: one block per (b,v) ----------------

__launch_bounds__(256, 1)
__global__ void gpnn_main(const float* __restrict__ edge, const float* __restrict__ node,
                          const float* __restrict__ b_gates, const _Float16* __restrict__ WgT,
                          const float* __restrict__ W_lout, const float* __restrict__ b_lout,
                          const _Float16* __restrict__ WmbT, const float* __restrict__ P,
                          const _Float16* __restrict__ WmgT, const float* __restrict__ b_mrg,
                          const float* __restrict__ ln_g, const float* __restrict__ ln_b,
                          float* __restrict__ out) {
    extern __shared__ char smem[];
    _Float16* Eh  = (_Float16*)(smem + OFF_E);
    _Float16* Hh  = (_Float16*)(smem + OFF_H);
    _Float16* MVh = (_Float16*)(smem + OFF_MV);
    float*    Mf  = (float*)(smem + OFF_M);
    float*    Wt  = (float*)(smem + OFF_WT);
    float*    St  = (float*)(smem + OFF_STAT);
    float*    MuR = (float*)(smem + OFF_MUR);

    const int tid = threadIdx.x;
    const int wave = tid >> 5;
    const int lane = tid & 31;
    const int halfsel = lane >> 4;
    const int l16 = lane & 15;
    const int bv = blockIdx.x;           // b*64 + v
    const int b64 = bv & ~63;            // b*64

    const v8f zero8 = {0.f, 0.f, 0.f, 0.f, 0.f, 0.f, 0.f, 0.f};

    // ---- stage edge slice [64][768] -> LDS f16 ----
    const float4* esrc = (const float4*)(edge + (size_t)bv * (64 * DD));
    for (int i = 0; i < 48; ++i) {
        float4 v = esrc[i * 256 + tid];
        v4h h4 = { (_Float16)v.x, (_Float16)v.y, (_Float16)v.z, (_Float16)v.w };
        *(v4h*)(Eh + (i * 256 + tid) * 4) = h4;
    }
    __syncthreads();

    // ---- phase 1: gates (i,g,o share A-fragments; f-gate unused) -> h f16 ----
    for (int g = 0; g < 2; ++g) {
        const int ct = wave + g * 8;                   // 0..15 column-tile within H
        const int n_i = ct * 16, n_g = 512 + ct * 16, n_o = DD + ct * 16;
        v8f ai[4] = {zero8, zero8, zero8, zero8};
        v8f ag[4] = {zero8, zero8, zero8, zero8};
        v8f ao[4] = {zero8, zero8, zero8, zero8};
        const _Float16* Bi = WgT + (size_t)(n_i + l16) * DD + 16 * halfsel;
        const _Float16* Bg = WgT + (size_t)(n_g + l16) * DD + 16 * halfsel;
        const _Float16* Bo = WgT + (size_t)(n_o + l16) * DD + 16 * halfsel;
        for (int kb = 0; kb < DD; kb += 32) {
            v16h bi = *(const v16h*)(Bi + kb);
            v16h bg = *(const v16h*)(Bg + kb);
            v16h bo = *(const v16h*)(Bo + kb);
#pragma unroll
            for (int mt = 0; mt < 4; ++mt) {
                v16h a = load_afrag(Eh, mt * 16 + l16, kb, halfsel);
                ai[mt] = __builtin_amdgcn_wmma_f32_16x16x32_f16(false, a, false, bi, (short)0, ai[mt], false, false);
                ag[mt] = __builtin_amdgcn_wmma_f32_16x16x32_f16(false, a, false, bg, (short)0, ag[mt], false, false);
                ao[mt] = __builtin_amdgcn_wmma_f32_16x16x32_f16(false, a, false, bo, (short)0, ao[mt], false, false);
            }
        }
        const float bi_s = b_gates[n_i + l16];
        const float bg_s = b_gates[n_g + l16];
        const float bo_s = b_gates[n_o + l16];
#pragma unroll
        for (int mt = 0; mt < 4; ++mt)
#pragma unroll
            for (int r = 0; r < 8; ++r) {
                float c = sigmoidf_(ai[mt][r] + bi_s) * tanhf_(ag[mt][r] + bg_s);
                float hv = sigmoidf_(ao[mt][r] + bo_s) * tanhf_(c);
                int row = mt * 16 + r + 8 * halfsel;
                Hh[row * HH + ct * 16 + l16] = (_Float16)hv;
            }
    }
    __syncthreads();

    // ---- edge weight: sigmoid(h . W_lout + b) (64 rows) ----
    if (tid < 64) {
        float s = 0.f;
        const _Float16* hr = Hh + tid * HH;
        for (int k = 0; k < HH; ++k) s += (float)hr[k] * W_lout[k];
        Wt[tid] = sigmoidf_(s + b_lout[0]);
    }

    // ---- phase 2: mv = E @ W_msg_bot + P[b,w] -> LDS f16 ----
    for (int j = 0; j < 6; ++j) {
        const int n0 = (wave + j * 8) * 16;
        v8f acc[4] = {zero8, zero8, zero8, zero8};
        const _Float16* Bp = WmbT + (size_t)(n0 + l16) * DD + 16 * halfsel;
        for (int kb = 0; kb < DD; kb += 32) {
            v16h bf = *(const v16h*)(Bp + kb);
#pragma unroll
            for (int mt = 0; mt < 4; ++mt) {
                v16h a = load_afrag(Eh, mt * 16 + l16, kb, halfsel);
                acc[mt] = __builtin_amdgcn_wmma_f32_16x16x32_f16(false, a, false, bf, (short)0, acc[mt], false, false);
            }
        }
        const int d = n0 + l16;
#pragma unroll
        for (int mt = 0; mt < 4; ++mt)
#pragma unroll
            for (int r = 0; r < 8; ++r) {
                int row = mt * 16 + r + 8 * halfsel;              // row == neighbor index w
                float v = acc[mt][r] + P[(size_t)(b64 + row) * DD + d];
                MVh[row * DD + d] = (_Float16)v;
            }
    }
    __syncthreads();

    // ---- phase 3: m = mv @ W_mrg + b_mrg -> LDS f32 (overlaps freed E/H) ----
    for (int j = 0; j < 6; ++j) {
        const int n0 = (wave + j * 8) * 16;
        v8f acc[4] = {zero8, zero8, zero8, zero8};
        const _Float16* Bp = WmgT + (size_t)(n0 + l16) * DD + 16 * halfsel;
        for (int kb = 0; kb < DD; kb += 32) {
            v16h bf = *(const v16h*)(Bp + kb);
#pragma unroll
            for (int mt = 0; mt < 4; ++mt) {
                v16h a = load_afrag(MVh, mt * 16 + l16, kb, halfsel);
                acc[mt] = __builtin_amdgcn_wmma_f32_16x16x32_f16(false, a, false, bf, (short)0, acc[mt], false, false);
            }
        }
        const int d = n0 + l16;
        const float bm = b_mrg[d];
#pragma unroll
        for (int mt = 0; mt < 4; ++mt)
#pragma unroll
            for (int r = 0; r < 8; ++r) {
                int row = mt * 16 + r + 8 * halfsel;
                Mf[row * DD + d] = acc[mt][r] + bm;
            }
    }
    __syncthreads();

    // ---- LayerNorm stats (4 partials per row) ----
    {
        int row = tid >> 2, part = tid & 3;
        float s = 0.f, s2 = 0.f;
        const float* mr = Mf + row * DD + part * 192;
        for (int k = 0; k < 192; ++k) { float v = mr[k]; s += v; s2 += v * v; }
        St[(row * 4 + part) * 2 + 0] = s;
        St[(row * 4 + part) * 2 + 1] = s2;
    }
    __syncthreads();
    if (tid < 64) {
        float s = 0.f, s2 = 0.f;
        for (int p = 0; p < 4; ++p) { s += St[(tid * 4 + p) * 2]; s2 += St[(tid * 4 + p) * 2 + 1]; }
        float mu = s * (1.0f / 768.0f);
        float var = s2 * (1.0f / 768.0f) - mu * mu;
        MuR[tid * 2 + 0] = mu;
        MuR[tid * 2 + 1] = rsqrtf(var + 1e-12f);
    }
    __syncthreads();

    // ---- LN + erf GELU + weighted neighbor sum + residual ----
    for (int j = 0; j < 3; ++j) {
        int d = tid + j * 256;
        float g = ln_g[d], bb = ln_b[d];
        float acc = 0.f;
        for (int r = 0; r < 64; ++r) {
            float x = g * ((Mf[r * DD + d] - MuR[r * 2]) * MuR[r * 2 + 1]) + bb;
            acc = fmaf(Wt[r], gelu_erf_(x), acc);
        }
        out[(size_t)bv * DD + d] = node[(size_t)bv * DD + d] + acc;
    }
}

extern "C" void kernel_launch(void* const* d_in, const int* in_sizes, int n_in,
                              void* d_out, int out_size, void* d_ws, size_t ws_size,
                              hipStream_t stream) {
    const float* edge    = (const float*)d_in[0];
    const float* node    = (const float*)d_in[1];
    const float* W_gates = (const float*)d_in[2];
    const float* b_gates = (const float*)d_in[3];
    const float* W_lout  = (const float*)d_in[4];
    const float* b_lout  = (const float*)d_in[5];
    const float* W_msg   = (const float*)d_in[6];
    const float* b_msg   = (const float*)d_in[7];
    const float* W_mrg   = (const float*)d_in[8];
    const float* b_mrg   = (const float*)d_in[9];
    const float* ln_g    = (const float*)d_in[10];
    const float* ln_b    = (const float*)d_in[11];
    float* out = (float*)d_out;

    char* ws = (char*)d_ws;
    _Float16* WgT  = (_Float16*)(ws + WS_WGT);
    _Float16* WmbT = (_Float16*)(ws + WS_WMB);
    _Float16* WmgT = (_Float16*)(ws + WS_WMG);
    float*    Pbuf = (float*)(ws + WS_P);

    prep_weights<<<7680, 256, 0, stream>>>(W_gates, W_msg, W_mrg, WgT, WmbT, WmgT);
    prep_P<<<512, 256, 0, stream>>>(node, W_msg, b_msg, Pbuf);

    hipFuncSetAttribute((const void*)gpnn_main, hipFuncAttributeMaxDynamicSharedMemorySize, SMEM_SZ);
    gpnn_main<<<512, 256, SMEM_SZ, stream>>>(edge, node, b_gates, WgT, W_lout, b_lout,
                                             WmbT, Pbuf, WmgT, b_mrg, ln_g, ln_b, out);
}
